// SAEFactorizer_65575560675384
// MI455X (gfx1250) — matrix operand order repaired and checked by reading, not compiled, not run on hardware
//
#include <hip/hip_runtime.h>

// ---------------------------------------------------------------------------
// Sizes (compile-time constants from the reference)
// ---------------------------------------------------------------------------
#define BB 4096
#define DD 768
#define FF 24576
#define KK 32

typedef __bf16 bf16;
typedef __attribute__((ext_vector_type(16))) __bf16    v16bf;
typedef __attribute__((ext_vector_type(8)))  __bf16    v8bf;
typedef __attribute__((ext_vector_type(8)))  float     v8f;
typedef __attribute__((ext_vector_type(4)))  float     v4f;
typedef __attribute__((ext_vector_type(8)))  _Float16  v8h;

// ---------------------------------------------------------------------------
// Kernel 1: xc = (x - b_pre) converted to bf16, row-major [B][D]
// ---------------------------------------------------------------------------
__global__ __launch_bounds__(256)
void k_xc_convert(const float* __restrict__ x, const float* __restrict__ b_pre,
                  bf16* __restrict__ xc) {
    int i = blockIdx.x * 256 + threadIdx.x;
    if (i < BB * DD) {
        int col = i % DD;
        xc[i] = (bf16)(x[i] - b_pre[col]);
    }
}

// ---------------------------------------------------------------------------
// Kernel 2: W_enc [D][F] f32  ->  W_encT [F][D] bf16 (LDS-tiled transpose)
// ---------------------------------------------------------------------------
__global__ __launch_bounds__(256)
void k_transpose_convert(const float* __restrict__ W, bf16* __restrict__ wT) {
    __shared__ float tile[32][33];
    int tx = threadIdx.x;           // 0..31
    int ty = threadIdx.y;           // 0..7
    int f0 = blockIdx.x * 32;       // F tile origin
    int d0 = blockIdx.y * 32;       // D tile origin
    #pragma unroll
    for (int j = 0; j < 32; j += 8)
        tile[ty + j][tx] = W[(size_t)(d0 + ty + j) * FF + f0 + tx];
    __syncthreads();
    #pragma unroll
    for (int j = 0; j < 32; j += 8)
        wT[(size_t)(f0 + ty + j) * DD + d0 + tx] = (bf16)tile[tx][ty + j];
}

// ---------------------------------------------------------------------------
// Kernel 3: pre_acts = xc @ W_enc  via v_wmma_f32_16x16x32_bf16
// Workgroup: 256 threads = 8 waves arranged 2(M) x 4(N).
// WG tile: 64(M) x 256(N). Wave tile: 32(M) x 64(N) = 2x4 WMMA frags.
// Result stored as f16 (halves intermediate HBM traffic; f16's 2^-11 error is
// below the bf16 GEMM noise floor and pre_acts is O(1)-scaled -> no overflow).
// ---------------------------------------------------------------------------
__device__ __forceinline__ v16bf load_a_frag(const bf16* __restrict__ rowBase,
                                             int kb, int lane) {
    int m    = lane & 15;
    int koff = (lane >> 4) * 16;
    const bf16* p = rowBase + (size_t)m * DD + kb + koff;
    v8bf lo = *(const v8bf*)(p);        // K = kb+koff   .. +7   (VGPR 0-3)
    v8bf hi = *(const v8bf*)(p + 8);    // K = kb+8+koff .. +7   (VGPR 4-7)
    return __builtin_shufflevector(lo, hi, 0,1,2,3,4,5,6,7,8,9,10,11,12,13,14,15);
}

__device__ __forceinline__ v16bf load_b_frag(const bf16* __restrict__ wT,
                                             int n0, int kb, int lane) {
    int n  = n0 + (lane & 15);
    int kk = kb + (lane >> 4) * 16;     // lanes 0-15: K=kb..+15, 16-31: +16..+31
    return *(const v16bf*)(wT + (size_t)n * DD + kk);
}

__global__ __launch_bounds__(256)
void k_gemm_enc(const bf16* __restrict__ xc, const bf16* __restrict__ wT,
                _Float16* __restrict__ pre) {
    const int lane = threadIdx.x & 31;
    const int wave = threadIdx.x >> 5;
    const int waveM = wave & 1;         // 0..1
    const int waveN = wave >> 1;        // 0..3
    const int M0 = blockIdx.y * 64 + waveM * 32;
    const int N0 = blockIdx.x * 256 + waveN * 64;

    v8f acc[2][4];
    #pragma unroll
    for (int i = 0; i < 2; ++i)
        #pragma unroll
        for (int j = 0; j < 4; ++j)
            acc[i][j] = (v8f){0.f,0.f,0.f,0.f,0.f,0.f,0.f,0.f};

    const bf16* aBase0 = xc + (size_t)M0 * DD;
    const bf16* aBase1 = xc + (size_t)(M0 + 16) * DD;

    for (int kb = 0; kb < DD; kb += 32) {
        // prefetch next K-slab into caches (global_prefetch_b8)
        if (kb + 32 < DD) {
            __builtin_prefetch(wT + (size_t)(N0 + (lane & 15)) * DD + kb + 32, 0, 1);
            __builtin_prefetch(aBase0 + (size_t)(lane & 15) * DD + kb + 32, 0, 1);
        }
        v16bf a0 = load_a_frag(aBase0, kb, lane);
        v16bf a1 = load_a_frag(aBase1, kb, lane);
        v16bf b0 = load_b_frag(wT, N0 +  0, kb, lane);
        v16bf b1 = load_b_frag(wT, N0 + 16, kb, lane);
        v16bf b2 = load_b_frag(wT, N0 + 32, kb, lane);
        v16bf b3 = load_b_frag(wT, N0 + 48, kb, lane);

        acc[0][0] = __builtin_amdgcn_wmma_f32_16x16x32_bf16(false, a0, false, b0, (short)0, acc[0][0], false, false);
        acc[0][1] = __builtin_amdgcn_wmma_f32_16x16x32_bf16(false, a0, false, b1, (short)0, acc[0][1], false, false);
        acc[0][2] = __builtin_amdgcn_wmma_f32_16x16x32_bf16(false, a0, false, b2, (short)0, acc[0][2], false, false);
        acc[0][3] = __builtin_amdgcn_wmma_f32_16x16x32_bf16(false, a0, false, b3, (short)0, acc[0][3], false, false);
        acc[1][0] = __builtin_amdgcn_wmma_f32_16x16x32_bf16(false, a1, false, b0, (short)0, acc[1][0], false, false);
        acc[1][1] = __builtin_amdgcn_wmma_f32_16x16x32_bf16(false, a1, false, b1, (short)0, acc[1][1], false, false);
        acc[1][2] = __builtin_amdgcn_wmma_f32_16x16x32_bf16(false, a1, false, b2, (short)0, acc[1][2], false, false);
        acc[1][3] = __builtin_amdgcn_wmma_f32_16x16x32_bf16(false, a1, false, b3, (short)0, acc[1][3], false, false);
    }

    // C/D layout: VGPR r -> row M0 + i*16 + (lane>=16 ? 8 : 0) + r; col = N0 + j*16 + lane%16
    #pragma unroll
    for (int i = 0; i < 2; ++i) {
        int mbase = M0 + i * 16 + (lane >> 4) * 8;
        #pragma unroll
        for (int j = 0; j < 4; ++j) {
            int n = N0 + j * 16 + (lane & 15);
            _Float16* outp = pre + (size_t)mbase * FF + n;
            #pragma unroll
            for (int r = 0; r < 8; ++r)
                outp[(size_t)r * FF] = (_Float16)acc[i][j][r];
        }
    }
}

// ---------------------------------------------------------------------------
// Kernel 4: per-row top-K (radix select over float-ordered keys), dense
// sparsified acts row written once (NT), plus sparse decoder x_hat row.
// One 256-thread workgroup per row; row staged f16->f32 in dynamic LDS.
// ---------------------------------------------------------------------------
__device__ __forceinline__ unsigned f2key(float v) {
    unsigned u = __float_as_uint(v);
    return (u & 0x80000000u) ? ~u : (u | 0x80000000u);   // monotonic: bigger key = bigger float
}

__global__ __launch_bounds__(256)
void k_topk_decode(const _Float16* __restrict__ pre,  // [B][F] f16 pre_acts
                   float* __restrict__ acts,          // [B][F] output (written once)
                   const float* __restrict__ Wdec,    // [F][D]
                   const float* __restrict__ b_pre,   // [D]
                   float* __restrict__ xhat) {        // [B][D]
    extern __shared__ unsigned char smem_raw[];
    float*    srow = (float*)smem_raw;                 // FF floats
    unsigned* hist = (unsigned*)(srow + FF);           // 256
    float*    selV = (float*)(hist + 256);             // KK
    int*      selI = (int*)(selV + KK);                // KK
    unsigned* ctr  = (unsigned*)(selI + KK);           // [0]=eqTaken [1]=nsel

    const int tid = threadIdx.x;
    const int row = blockIdx.x;
    const _Float16* prow = pre + (size_t)row * FF;
    float* arow = acts + (size_t)row * FF;

    // stage row in LDS, converting f16 -> f32 (8-wide vector loads)
    for (int i = tid * 8; i < FF; i += 256 * 8) {
        v8h h = *(const v8h*)(prow + i);
        #pragma unroll
        for (int j = 0; j < 8; ++j) srow[i + j] = (float)h[j];
    }
    if (tid < KK) { selV[tid] = 0.f; selI[tid] = 0; }
    if (tid == 0) { ctr[0] = 0u; ctr[1] = 0u; }
    __syncthreads();

    // 4-pass MSB-first radix select for the KK-th largest key
    unsigned prefix = 0;
    int remaining = KK;
    for (int shift = 24; shift >= 0; shift -= 8) {
        hist[tid] = 0u;
        __syncthreads();
        unsigned himask = (shift == 24) ? 0u : (0xFFFFFFFFu << (shift + 8));
        for (int i = tid; i < FF; i += 256) {
            unsigned key = f2key(srow[i]);
            if ((key & himask) == prefix)
                atomicAdd(&hist[(key >> shift) & 255u], 1u);
        }
        __syncthreads();
        int cum = 0, bsel = 0;
        for (int b = 255; b >= 0; --b) {           // every thread scans identically
            int c = (int)hist[b];
            if (cum + c >= remaining) { bsel = b; break; }
            cum += c;
        }
        remaining -= cum;                          // still needed within selected bucket
        prefix |= ((unsigned)bsel) << shift;
        __syncthreads();                           // hist reused next pass
    }
    const unsigned thresh = prefix;

    // write sparsified row once (NT, 4-wide); collect the KK winners
    for (int i0 = tid * 4; i0 < FF; i0 += 256 * 4) {
        v4f o;
        #pragma unroll
        for (int j = 0; j < 4; ++j) {
            int i = i0 + j;
            float v = srow[i];
            unsigned key = f2key(v);
            bool keep = false;
            if (key > thresh) keep = true;
            else if (key == thresh) {
                unsigned pos = atomicAdd(&ctr[0], 1u); // tie-break among equal keys
                keep = (pos < (unsigned)remaining);
            }
            o[j] = keep ? v : 0.0f;
            if (keep) {
                unsigned s = atomicAdd(&ctr[1], 1u);
                if (s < (unsigned)KK) { selV[s] = v; selI[s] = i; }
            }
        }
        __builtin_nontemporal_store(o, (v4f*)(arow + i0));
    }
    __syncthreads();

    // sparse decoder: x_hat[row] = b_pre + sum_k selV[k] * Wdec[selI[k], :]
    for (int col = tid; col < DD; col += 256) {
        float s = b_pre[col];
        #pragma unroll 8
        for (int t = 0; t < KK; ++t)
            s += selV[t] * Wdec[(size_t)selI[t] * DD + col];
        xhat[(size_t)row * DD + col] = s;
    }
}

// ---------------------------------------------------------------------------
// Entry point
// ---------------------------------------------------------------------------
extern "C" void kernel_launch(void* const* d_in, const int* in_sizes, int n_in,
                              void* d_out, int out_size, void* d_ws, size_t ws_size,
                              hipStream_t stream) {
    const float* x     = (const float*)d_in[0];   // [B][D]
    const float* W_enc = (const float*)d_in[1];   // [D][F]
    const float* W_dec = (const float*)d_in[2];   // [F][D]
    const float* b_pre = (const float*)d_in[3];   // [D]
    // d_in[4] is k (==32), baked in at compile time

    float* xhat = (float*)d_out;                            // [B][D]
    float* acts = (float*)d_out + (size_t)BB * DD;          // [B][F]

    bf16*     xc    = (bf16*)d_ws;                          // [B][D]  (6.3 MB)
    bf16*     wT    = xc + (size_t)BB * DD;                 // [F][D]  (37.7 MB)
    _Float16* pre16 = (_Float16*)(wT + (size_t)FF * DD);    // [B][F]  (201 MB)

    // 1) centered input -> bf16
    k_xc_convert<<<(BB * DD + 255) / 256, 256, 0, stream>>>(x, b_pre, xc);

    // 2) W_enc -> transposed bf16 (makes WMMA B-fragments contiguous)
    {
        dim3 grid(FF / 32, DD / 32);
        dim3 block(32, 8);
        k_transpose_convert<<<grid, block, 0, stream>>>(W_enc, wT);
    }

    // 3) encoder GEMM -> pre_acts f16 in workspace
    {
        dim3 grid(FF / 256, BB / 64);
        k_gemm_enc<<<grid, 256, 0, stream>>>(xc, wT, pre16);
    }

    // 4) per-row top-K + single-pass acts write + sparse decoder
    {
        size_t smem = (size_t)(FF + 256 + KK + KK + 2) * 4;
        k_topk_decode<<<BB, 256, smem, stream>>>(pre16, acts, W_dec, b_pre, xhat);
    }
}
